// SelfAttentionWithMemory_77266461655813
// MI455X (gfx1250) — matrix operand order repaired and checked by reading, not compiled, not run on hardware
//
#include <hip/hip_runtime.h>

// ---- CDNA5 (gfx1250) types -------------------------------------------------
typedef __bf16 bf16;
typedef __attribute__((ext_vector_type(16))) __bf16 v16bf;
typedef __attribute__((ext_vector_type(8)))  __bf16 v8bf;
typedef __attribute__((ext_vector_type(8)))  float  v8f;
typedef __attribute__((ext_vector_type(4))) unsigned int u32x4;
typedef __attribute__((ext_vector_type(8))) int          i32x8;
typedef __attribute__((ext_vector_type(4))) int          i32x4;

#define WMMA_BF16(A, Bm, C) \
  __builtin_amdgcn_wmma_f32_16x16x32_bf16(false, (A), false, (Bm), (short)0, (C), false, false)

// ---- problem constants -----------------------------------------------------
#define NUM_HEADS 12
#define HEAD_DIM  64
#define B_   8
#define S_   1024
#define D_   768
#define M_   8
#define SM_  1032              // S + M (K/V sequence length)
#define KPAD 1056              // SM padded to multiple of 32 (33 * 32)
#define KSTEPS (KPAD / 32)     // 33 attention k-steps
#define ROWS (B_ * KPAD)       // 8448 padded token rows
#define NEG_INF (-3.0e38f)

static __device__ __forceinline__ v8f zero8() {
  v8f z = {0.f, 0.f, 0.f, 0.f, 0.f, 0.f, 0.f, 0.f};
  return z;
}

// A fragment (16x32 bf16): two 16B chunks per lane at 2*sel8 / 2*sel8+8
static __device__ __forceinline__ v16bf load_frag_a(const bf16* row, int sel8) {
  v8bf c0 = *(const v8bf*)(row + 2 * sel8);
  v8bf c1 = *(const v8bf*)(row + 2 * sel8 + 8);
  v16bf a;
#pragma unroll
  for (int i = 0; i < 8; ++i) { a[i] = c0[i]; a[i + 8] = c1[i]; }
  return a;
}

// ---- Tensor Data Mover: 2-D tile load Global -> LDS ------------------------
// D# packing per CDNA5 ISA (bf16 elements, data_size=1).
static __device__ __forceinline__ void tdm_load_2d(
    unsigned lds_byte_off, const void* gptr,
    unsigned tile_d0, unsigned tile_d1,          // tile (elems, rows)
    unsigned tensor_d0, unsigned tensor_d1,      // tensor extent for OOB
    unsigned stride_elems) {                     // dim0 stride (elems)
  unsigned long long ga = (unsigned long long)gptr;
  u32x4 g0;
  g0[0] = 1u;                                        // count=1, user descriptor
  g0[1] = lds_byte_off;                              // lds_addr [63:32]
  g0[2] = (unsigned)(ga & 0xFFFFFFFFu);              // global_addr lo
  g0[3] = (unsigned)((ga >> 32) & 0x01FFFFFFu)       // global_addr [56:32]
        | (2u << 30);                                // type = 2 ("image")
  i32x8 g1;
  g1[0] = (int)(1u << 16);                           // data_size=1 (2 bytes)
  g1[1] = (int)((tensor_d0 & 0xFFFFu) << 16);        // tensor_dim0[15:0] @63:48
  g1[2] = (int)(((tensor_d0 >> 16) & 0xFFFFu) |      // tensor_dim0[31:16]
                ((tensor_d1 & 0xFFFFu) << 16));      // tensor_dim1[15:0]
  g1[3] = (int)(((tensor_d1 >> 16) & 0xFFFFu) |      // tensor_dim1[31:16]
                ((tile_d0 & 0xFFFFu) << 16));        // tile_dim0 @127:112
  g1[4] = (int)(tile_d1 & 0xFFFFu);                  // tile_dim1; tile_dim2=0
  g1[5] = (int)stride_elems;                         // tensor_dim0_stride lo32
  g1[6] = 0;                                         // stride hi / dim1_stride
  g1[7] = 0;
  i32x4 z4 = {0, 0, 0, 0};
#if defined(__clang_major__) && (__clang_major__ >= 23)
  i32x8 z8 = {0, 0, 0, 0, 0, 0, 0, 0};
  __builtin_amdgcn_tensor_load_to_lds(g0, g1, z4, z4, z8, 0);
#else
  __builtin_amdgcn_tensor_load_to_lds(g0, g1, z4, z4, 0);
#endif
}

static __device__ __forceinline__ unsigned lds_off(const void* p) {
  return (unsigned)(unsigned long long)p;  // LDS offset = addr[31:0]
}

// ---- pack concat(hidden, memory) -> bf16, zero-padded to KPAD rows ---------
__global__ __launch_bounds__(256) void pack_x_kernel(
    const float* __restrict__ hs, const float* __restrict__ mem,
    bf16* __restrict__ X) {
  int idx = blockIdx.x * blockDim.x + threadIdx.x;
  if (idx >= ROWS * D_) return;
  int d   = idx % D_;
  int row = idx / D_;
  int t = row % KPAD;
  int b = row / KPAD;
  float v = 0.f;
  if (t < S_)       v = hs[((size_t)(b * S_ + t)) * D_ + d];
  else if (t < SM_) v = mem[(size_t)(t - S_) * D_ + d];
  X[idx] = (bf16)v;
}

__global__ __launch_bounds__(256) void pack_w_kernel(
    const float* __restrict__ W, bf16* __restrict__ Wb) {
  int idx = blockIdx.x * blockDim.x + threadIdx.x;
  if (idx < D_ * D_) Wb[idx] = (bf16)W[idx];
}

// ---- projection GEMM: Y(rows x 768) = X @ W^T + bias -----------------------
// One wave: 32-row x 64-feature tile (8 accumulators), K-loop step 32 with
// one-deep register prefetch; unroll-by-2 lets the compiler modulo-rename the
// ping-pong fragment registers (no v_mov rotation, fewer hazard NOPs).
__global__ __launch_bounds__(128) void proj_gemm_kernel(
    const bf16* __restrict__ X, const bf16* __restrict__ W,
    const float* __restrict__ bias, bf16* __restrict__ out, int transposed) {
  const int lane = threadIdx.x & 31;
  const int wv   = threadIdx.x >> 5;
  const int ntn  = D_ / 64;                        // 12 feature tiles
  int tile = blockIdx.x * 4 + wv;
  if (tile >= (ROWS / 32) * ntn) return;
  const int rt = tile / ntn;                       // 32-row tile
  const int nt = tile % ntn;                       // 64-feature tile
  const int n15   = lane & 15;
  const int sel8  = (lane < 16) ? 0 : 8;
  const int sel16 = (lane < 16) ? 0 : 16;

  const bf16* xr0 = X + (size_t)(rt * 32 + n15) * D_;
  const bf16* xr1 = xr0 + (size_t)16 * D_;
  const bf16* wr  = W + (size_t)(nt * 64 + n15) * D_ + sel16;

  v8f acc0[4], acc1[4];
#pragma unroll
  for (int f = 0; f < 4; ++f) { acc0[f] = zero8(); acc1[f] = zero8(); }

  v16bf a0 = load_frag_a(xr0, sel8);
  v16bf a1 = load_frag_a(xr1, sel8);
  v16bf b0 = *(const v16bf*)(wr);
  v16bf b1 = *(const v16bf*)(wr + 16 * D_);
  v16bf b2 = *(const v16bf*)(wr + 32 * D_);
  v16bf b3 = *(const v16bf*)(wr + 48 * D_);

#pragma unroll 2
  for (int k0 = 0; k0 < D_; k0 += 32) {
    v16bf na0 = a0, na1 = a1, nb0 = b0, nb1 = b1, nb2 = b2, nb3 = b3;
    const int kn = k0 + 32;
    if (kn < D_) {  // prefetch next iteration's fragments first
      na0 = load_frag_a(xr0 + kn, sel8);
      na1 = load_frag_a(xr1 + kn, sel8);
      nb0 = *(const v16bf*)(wr + kn);
      nb1 = *(const v16bf*)(wr + 16 * D_ + kn);
      nb2 = *(const v16bf*)(wr + 32 * D_ + kn);
      nb3 = *(const v16bf*)(wr + 48 * D_ + kn);
    }
    acc0[0] = WMMA_BF16(a0, b0, acc0[0]);
    acc0[1] = WMMA_BF16(a0, b1, acc0[1]);
    acc0[2] = WMMA_BF16(a0, b2, acc0[2]);
    acc0[3] = WMMA_BF16(a0, b3, acc0[3]);
    acc1[0] = WMMA_BF16(a1, b0, acc1[0]);
    acc1[1] = WMMA_BF16(a1, b1, acc1[1]);
    acc1[2] = WMMA_BF16(a1, b2, acc1[2]);
    acc1[3] = WMMA_BF16(a1, b3, acc1[3]);
    a0 = na0; a1 = na1; b0 = nb0; b1 = nb1; b2 = nb2; b3 = nb3;
  }

#pragma unroll
  for (int half = 0; half < 2; ++half) {
    v8f* acc = half ? acc1 : acc0;
    const int rbase = rt * 32 + half * 16;
    if (!transposed) {
#pragma unroll
      for (int f = 0; f < 4; ++f) {
        int feat = nt * 64 + f * 16 + n15;
        float bb = bias[feat];
#pragma unroll
        for (int j = 0; j < 8; ++j) {
          int r = rbase + j + sel8;
          out[(size_t)r * D_ + feat] = (bf16)(acc[f][j] + bb);
        }
      }
    } else {  // V^T layout: (b, feature, token)
      int b  = rbase / KPAD;
      int t0 = rbase % KPAD + sel8;
#pragma unroll
      for (int f = 0; f < 4; ++f) {
        int feat = nt * 64 + f * 16 + n15;
        float bb = bias[feat];
        v8bf pk;
#pragma unroll
        for (int j = 0; j < 8; ++j) pk[j] = (bf16)(acc[f][j] + bb);
        *(v8bf*)(out + ((size_t)(b * D_ + feat)) * KPAD + t0) = pk;
      }
    }
  }
}

// ---- flash attention: block = (b, h, 64 queries); 4 waves share K/V tiles --
// Wave 0 drives double-buffered TDM tensor_load_to_lds for the 32x64 K tile
// and 64x32 V^T tile; all waves sync via workgroup barrier + s_wait_tensorcnt.
__global__ __launch_bounds__(128) void flash_attn_kernel(
    const bf16* __restrict__ Q, const bf16* __restrict__ K,
    const bf16* __restrict__ Vt, float* __restrict__ out) {
  __shared__ __align__(128) bf16 kbuf[2][32 * 64];   // 8 KB
  __shared__ __align__(128) bf16 vbuf[2][64 * 32];   // 8 KB
  __shared__ __align__(16)  bf16 pbuf[4][16 * 32];   // 4 KB (per-wave P tile)
  const int lane = threadIdx.x & 31;
  const int wv   = threadIdx.x >> 5;
  const int blk  = blockIdx.x;                       // 1536 blocks
  const int qg = blk % (S_ / 64);                    // 64-query group
  const int h  = (blk / (S_ / 64)) % NUM_HEADS;
  const int b  = blk / ((S_ / 64) * NUM_HEADS);
  const int qt = qg * 4 + wv;                        // this wave's 16-query tile
  const int n15   = lane & 15;
  const int sel8  = (lane < 16) ? 0 : 8;
  const int sel16 = (lane < 16) ? 0 : 16;
  bf16* pl = pbuf[wv];

  const bf16* kbase = K  + ((size_t)(b * KPAD)) * D_ + h * HEAD_DIM;
  const bf16* vbase = Vt + ((size_t)(b * D_ + h * HEAD_DIM)) * KPAD;

  // Q A-fragments for d=0..31 (a0) and d=32..63 (a1)
  const bf16* qrow = Q + ((size_t)(b * KPAD + qt * 16 + n15)) * D_ + h * HEAD_DIM;
  v16bf a0 = load_frag_a(qrow, sel8);
  v16bf a1 = load_frag_a(qrow + 32, sel8);

  float mrun[8], lrun[8];
#pragma unroll
  for (int j = 0; j < 8; ++j) { mrun[j] = NEG_INF; lrun[j] = 0.f; }
  v8f acc[4];
#pragma unroll
  for (int f = 0; f < 4; ++f) acc[f] = zero8();

  // prologue: DMA step-0 tiles into buffer 0
  if (wv == 0) {
    tdm_load_2d(lds_off(kbuf[0]), kbase, 64, 32, 64, 32, D_);
    tdm_load_2d(lds_off(vbuf[0]), vbase, 32, 64, 32, 64, KPAD);
  }

#pragma unroll 1
  for (int step = 0; step < KSTEPS; ++step) {
    const int kt0 = step * 32;
    const int cur = step & 1;
    // barrier A: all waves finished reading buffer (cur^1) from step-1
    asm volatile("s_wait_dscnt 0x0" ::: "memory");
    __syncthreads();
    if (wv == 0) {
      if (step + 1 < KSTEPS) {
        const int kt1 = kt0 + 32;
        tdm_load_2d(lds_off(kbuf[cur ^ 1]), kbase + (size_t)kt1 * D_,
                    64, 32, 64, 32, D_);
        tdm_load_2d(lds_off(vbuf[cur ^ 1]), vbase + kt1,
                    32, 64, 32, 64, KPAD);
        __builtin_amdgcn_s_wait_tensorcnt(2);   // step's own tiles done
      } else {
        __builtin_amdgcn_s_wait_tensorcnt(0);
      }
    }
    __syncthreads();   // barrier B: buffer `cur` ready for every wave
    asm volatile("" ::: "memory");

    const bf16* kt = kbuf[cur];
    const bf16* vt = vbuf[cur];

    // ---- scores: two 16-token subtiles, accumulate over Dh=64
    v8f s0 = zero8(), s1 = zero8();
    {
      v16bf kb;
      kb = *(const v16bf*)(kt + n15 * 64 + sel16);              s0 = WMMA_BF16(a0, kb, s0);
      kb = *(const v16bf*)(kt + n15 * 64 + 32 + sel16);         s0 = WMMA_BF16(a1, kb, s0);
      kb = *(const v16bf*)(kt + (16 + n15) * 64 + sel16);       s1 = WMMA_BF16(a0, kb, s1);
      kb = *(const v16bf*)(kt + (16 + n15) * 64 + 32 + sel16);  s1 = WMMA_BF16(a1, kb, s1);
    }
    // ---- scale + mask padded columns (token >= 1032)
    const bool msk0 = (kt0 + n15)      >= SM_;
    const bool msk1 = (kt0 + 16 + n15) >= SM_;
#pragma unroll
    for (int j = 0; j < 8; ++j) {
      s0[j] = msk0 ? NEG_INF : s0[j] * 0.125f;
      s1[j] = msk1 ? NEG_INF : s1[j] * 0.125f;
    }
    // ---- online softmax (row reductions across 16-lane groups)
    float tmax[8];
#pragma unroll
    for (int j = 0; j < 8; ++j) tmax[j] = fmaxf(s0[j], s1[j]);
#pragma unroll
    for (int xm = 8; xm >= 1; xm >>= 1)
#pragma unroll
      for (int j = 0; j < 8; ++j)
        tmax[j] = fmaxf(tmax[j], __shfl_xor(tmax[j], xm, 32));
    float corr[8];
#pragma unroll
    for (int j = 0; j < 8; ++j) {
      float mn = fmaxf(mrun[j], tmax[j]);
      corr[j] = __expf(mrun[j] - mn);
      mrun[j] = mn;
    }
#pragma unroll
    for (int j = 0; j < 8; ++j) {
      s0[j] = __expf(s0[j] - mrun[j]);
      s1[j] = __expf(s1[j] - mrun[j]);
    }
    float tsum[8];
#pragma unroll
    for (int j = 0; j < 8; ++j) tsum[j] = s0[j] + s1[j];
#pragma unroll
    for (int xm = 8; xm >= 1; xm >>= 1)
#pragma unroll
      for (int j = 0; j < 8; ++j) tsum[j] += __shfl_xor(tsum[j], xm, 32);
#pragma unroll
    for (int j = 0; j < 8; ++j) lrun[j] = lrun[j] * corr[j] + tsum[j];
#pragma unroll
    for (int f = 0; f < 4; ++f)
#pragma unroll
      for (int j = 0; j < 8; ++j) acc[f][j] *= corr[j];

    // ---- P: C-layout -> A-layout via per-wave LDS round trip
#pragma unroll
    for (int j = 0; j < 8; ++j) {
      int r = j + sel8;
      pl[r * 32 + n15]      = (bf16)s0[j];
      pl[r * 32 + 16 + n15] = (bf16)s1[j];
    }
    asm volatile("s_wait_dscnt 0x0" ::: "memory");
    v16bf pa;
    {
      v8bf c0 = *(const v8bf*)(pl + n15 * 32 + sel16);
      v8bf c1 = *(const v8bf*)(pl + n15 * 32 + sel16 + 8);
#pragma unroll
      for (int i = 0; i < 8; ++i) { pa[i] = c0[i]; pa[i + 8] = c1[i]; }
    }
    // ---- ctx += P (16x32) @ V (32x64) from the shared V^T tile
#pragma unroll
    for (int f = 0; f < 4; ++f) {
      v16bf vb = *(const v16bf*)(vt + (f * 16 + n15) * 32 + sel16);
      acc[f] = WMMA_BF16(pa, vb, acc[f]);
    }
  }

  // ---- epilogue: divide by row sum, store f32 (B, S, D)
#pragma unroll
  for (int f = 0; f < 4; ++f) {
#pragma unroll
    for (int j = 0; j < 8; ++j) {
      int q   = qt * 16 + j + sel8;
      int col = h * HEAD_DIM + f * 16 + n15;
      out[((size_t)(b * S_ + q)) * D_ + col] = acc[f][j] / lrun[j];
    }
  }
}

// ---- host launcher ----------------------------------------------------------
extern "C" void kernel_launch(void* const* d_in, const int* in_sizes, int n_in,
                              void* d_out, int out_size, void* d_ws, size_t ws_size,
                              hipStream_t stream) {
  const float* hs  = (const float*)d_in[0];
  const float* mem = (const float*)d_in[1];
  const float* Wq  = (const float*)d_in[2];
  const float* bq  = (const float*)d_in[3];
  const float* Wk  = (const float*)d_in[4];
  const float* bk  = (const float*)d_in[5];
  const float* Wv  = (const float*)d_in[6];
  const float* bv  = (const float*)d_in[7];
  float* out = (float*)d_out;

  // workspace carve-out (~55.5 MB total)
  char* ws = (char*)d_ws;
  size_t off = 0;
  auto carve = [&](size_t bytes) -> void* {
    void* p = ws + off;
    off += (bytes + 255) & ~(size_t)255;
    return p;
  };
  bf16* Xb  = (bf16*)carve((size_t)ROWS * D_ * 2);
  bf16* Wqb = (bf16*)carve((size_t)D_ * D_ * 2);
  bf16* Wkb = (bf16*)carve((size_t)D_ * D_ * 2);
  bf16* Wvb = (bf16*)carve((size_t)D_ * D_ * 2);
  bf16* Qb  = (bf16*)carve((size_t)ROWS * D_ * 2);
  bf16* Kb  = (bf16*)carve((size_t)ROWS * D_ * 2);
  bf16* Vtb = (bf16*)carve((size_t)ROWS * D_ * 2);
  if (off > ws_size) return;  // workspace too small; nothing safe to do

  // 1) pack inputs to bf16
  {
    int n = ROWS * D_;
    pack_x_kernel<<<(n + 255) / 256, 256, 0, stream>>>(hs, mem, Xb);
  }
  {
    int n = D_ * D_;
    pack_w_kernel<<<(n + 255) / 256, 256, 0, stream>>>(Wq, Wqb);
    pack_w_kernel<<<(n + 255) / 256, 256, 0, stream>>>(Wk, Wkb);
    pack_w_kernel<<<(n + 255) / 256, 256, 0, stream>>>(Wv, Wvb);
  }
  // 2) QKV projection GEMMs (WMMA bf16, 32x64 wave tiles, pipelined)
  {
    int tiles  = (ROWS / 32) * (D_ / 64);  // 3168
    int blocks = tiles / 4;                // 792 exact
    proj_gemm_kernel<<<blocks, 128, 0, stream>>>(Xb, Wqb, bq, Qb, 0);
    proj_gemm_kernel<<<blocks, 128, 0, stream>>>(Xb, Wkb, bk, Kb, 0);
    proj_gemm_kernel<<<blocks, 128, 0, stream>>>(Xb, Wvb, bv, Vtb, 1);
  }
  // 3) flash attention (WMMA bf16 + TDM double-buffered K/V tiles)
  {
    int blocks = B_ * NUM_HEADS * (S_ / 64);  // 1536
    flash_attn_kernel<<<blocks, 128, 0, stream>>>(Qb, Kb, Vtb, out);
  }
}
